// BeamDecoder_19877108646611
// MI455X (gfx1250) — matrix operand order repaired and checked by reading, not compiled, not run on hardware
//
#include <hip/hip_runtime.h>
#include <stdint.h>

#define VOCAB 128000
#define TLEN  512
#define BEAM  32
#define NSTEPS 30
#define HDIM  128
#define SOS   1

typedef __attribute__((ext_vector_type(16))) __bf16 v16bf;
typedef __attribute__((ext_vector_type(8)))  float  v8f;
typedef __attribute__((ext_vector_type(4)))  unsigned int v4u;

__device__ __forceinline__ unsigned short f2bf(float f) {
  unsigned int u = __float_as_uint(f);
  u += 0x7FFFu + ((u >> 16) & 1u);          // round-to-nearest-even
  return (unsigned short)(u >> 16);
}
__device__ __forceinline__ float sigf(float x) { return 1.0f / (1.0f + expf(-x)); }

// ---------------- one-time kernels ----------------

// W_cp fp32 -> bf16 (131MB -> 65.5MB; bf16 copy is L2-resident for all 30 steps)
__global__ void k_convert_wcp(const float* __restrict__ w, unsigned short* __restrict__ out) {
  int base = blockIdx.x * 1024 + threadIdx.x;      // 32000 blocks * 1024 = 32,768,000
#pragma unroll
  for (int j = 0; j < 4; ++j) out[base + j * 256] = f2bf(w[base + j * 256]);
}

__global__ void k_init(float* __restrict__ state0, float* __restrict__ scores,
                       int* __restrict__ prevw, int* __restrict__ seq) {
  int tid = threadIdx.x;
  for (int i = tid; i < 5 * BEAM * HDIM; i += 256) state0[i] = 0.0f;  // h1,c1,h2,c2,ctx
  if (tid < BEAM) { scores[tid] = 0.0f; prevw[tid] = SOS; }
  for (int i = tid; i < BEAM * (NSTEPS + 1); i += 256) seq[i] = SOS;
}

// ---------------- per-step small kernels ----------------

// x1 = [embedding[prev_words], context]  (32 x 256)
__global__ void k_build_x1(const int* __restrict__ prevw, const float* __restrict__ emb,
                           const float* __restrict__ ctx, float* __restrict__ x1) {
  int o = blockIdx.x * 256 + threadIdx.x;          // 8192
  int b = o >> 8, j = o & 255;
  x1[o] = (j < 128) ? emb[(size_t)prevw[b] * 128 + j] : ctx[b * 128 + (j - 128)];
}

// generic LSTM cell: one thread per (beam, hidden) output; gate order i,f,g,o
__global__ void k_lstm(const float* __restrict__ x, int xlen,
                       const float* __restrict__ hin, const float* __restrict__ cin,
                       const float* __restrict__ Wih, const float* __restrict__ Whh,
                       const float* __restrict__ bih, const float* __restrict__ bhh,
                       float* __restrict__ hout, float* __restrict__ cout) {
  int o = blockIdx.x * 256 + threadIdx.x;          // 4096
  int b = o >> 7, j = o & 127;
  float gi = bih[j]       + bhh[j];
  float gf = bih[j + 128] + bhh[j + 128];
  float gg = bih[j + 256] + bhh[j + 256];
  float go = bih[j + 384] + bhh[j + 384];
  const float* xb = x + b * xlen;
  for (int k = 0; k < xlen; ++k) {
    float xv = xb[k];
    gi += Wih[j * xlen + k] * xv;
    gf += Wih[(j + 128) * xlen + k] * xv;
    gg += Wih[(j + 256) * xlen + k] * xv;
    go += Wih[(j + 384) * xlen + k] * xv;
  }
  const float* hb = hin + b * 128;
  for (int k = 0; k < 128; ++k) {
    float hv = hb[k];
    gi += Whh[j * 128 + k] * hv;
    gf += Whh[(j + 128) * 128 + k] * hv;
    gg += Whh[(j + 256) * 128 + k] * hv;
    go += Whh[(j + 384) * 128 + k] * hv;
  }
  float c = sigf(gf) * cin[o] + sigf(gi) * tanhf(gg);
  cout[o] = c;
  hout[o] = sigf(go) * tanhf(c);
}

__global__ void k_query(const float* __restrict__ h2, const float* __restrict__ Wq,
                        const float* __restrict__ bq, float* __restrict__ query) {
  int o = blockIdx.x * 256 + threadIdx.x;          // 4096
  int b = o >> 7, q = o & 127;
  float acc = bq[q];
  for (int k = 0; k < 128; ++k) acc += Wq[q * 128 + k] * h2[b * 128 + k];
  query[o] = acc;
}

__global__ void k_energy(const float* __restrict__ query, const float* __restrict__ key,
                         float* __restrict__ energy) {
  int o = blockIdx.x * 256 + threadIdx.x;          // 16384
  int b = o >> 9, t = o & 511;
  float acc = 0.0f;
  for (int k = 0; k < 128; ++k) acc += query[b * 128 + k] * key[t * 128 + k];
  energy[o] = acc;
}

// masked softmax over T + context; also emits bf16 [h2, ctx] for the WMMA GEMM
__global__ void k_attn(const float* __restrict__ energy, const float* __restrict__ mask,
                       const float* __restrict__ vals, const float* __restrict__ h2,
                       float* __restrict__ ctx_out, unsigned short* __restrict__ xbf) {
  int b = blockIdx.x, tid = threadIdx.x;
  __shared__ float w[TLEN];
  __shared__ float red[256];
  __shared__ float sM, sS;
  float e0 = energy[b * TLEN + tid], e1 = energy[b * TLEN + 256 + tid];
  red[tid] = fmaxf(e0, e1); __syncthreads();
  for (int s = 128; s > 0; s >>= 1) { if (tid < s) red[tid] = fmaxf(red[tid], red[tid + s]); __syncthreads(); }
  if (tid == 0) sM = red[0]; __syncthreads();
  float w0 = mask[tid] * expf(e0 - sM);
  float w1 = mask[tid + 256] * expf(e1 - sM);
  w[tid] = w0; w[tid + 256] = w1;
  red[tid] = w0 + w1; __syncthreads();
  for (int s = 128; s > 0; s >>= 1) { if (tid < s) red[tid] += red[tid + s]; __syncthreads(); }
  if (tid == 0) sS = red[0]; __syncthreads();
  if (tid < 128) {
    float acc = 0.0f;
    for (int t = 0; t < TLEN; ++t) acc += w[t] * vals[t * 128 + tid];
    acc /= sS;
    ctx_out[b * 128 + tid] = acc;
    xbf[b * 256 + 128 + tid] = f2bf(acc);
    xbf[b * 256 + tid] = f2bf(h2[b * 128 + tid]);
  }
}

// ---------------- the big one: pred = x(32x256,bf16) @ Wcp^T(256x128000,bf16) + b ----------------
// block = 256 thr = 8 waves; wave w: beam-half = w&1, vocab sub-tile = w>>1 (16 cols);
// block covers 64 vocab cols; grid = 2000 blocks; K=256 as 8 x wmma 16x16x32 bf16.
__global__ void k_pred_wmma(const unsigned short* __restrict__ xb,
                            const unsigned short* __restrict__ wb,
                            const float* __restrict__ bias,
                            float* __restrict__ pred) {
  int wv = threadIdx.x >> 5, lane = threadIdx.x & 31;
  int beamHalf = wv & 1, nsub = wv >> 1;
  int l15 = lane & 15;
  int hi = lane >> 4;                                   // 0: lanes 0-15, 1: lanes 16-31
  int rowA = beamHalf * 16 + l15;                       // beam
  int colB = blockIdx.x * 64 + nsub * 16 + l15;         // vocab id
  // A layout (16-bit 16x32): lo lanes K{0-7,16-23}, hi lanes K{8-15,24-31}
  const unsigned short* aRow = xb + rowA * 256 + (hi ? 8 : 0);
  // B layout (16-bit 32x16): lane = column; lo lanes K{0-7,16-23}, hi lanes K{8-15,24-31} (2 K per VGPR)
  const unsigned short* bRow = wb + (size_t)colB * 256 + (hi ? 8 : 0);
  union Frag { v4u q[2]; v16bf v; };
  v8f acc = {};
#pragma unroll
  for (int kb = 0; kb < 256; kb += 32) {
    Frag A, B;
    A.q[0] = *(const v4u*)(aRow + kb);
    A.q[1] = *(const v4u*)(aRow + kb + 16);
    B.q[0] = *(const v4u*)(bRow + kb);
    B.q[1] = *(const v4u*)(bRow + kb + 16);
    acc = __builtin_amdgcn_wmma_f32_16x16x32_bf16(false, A.v, false, B.v,
                                                  (short)0, acc, false, false);
  }
  float bv = bias[colB];
  int mBase = beamHalf * 16 + (hi ? 8 : 0);             // C/D: elem r -> row r (+8 for hi lanes)
#pragma unroll
  for (int r = 0; r < 8; ++r)
    pred[(size_t)(mBase + r) * VOCAB + colB] = acc[r] + bv;
}

// ---------------- logsumexp (2-pass; pred is L2-resident) ----------------
__global__ void k_lse_part(const float* __restrict__ pred, float* __restrict__ pmax,
                           float* __restrict__ psum) {
  int beam = blockIdx.x / 125, chunk = blockIdx.x % 125;  // 4000 blocks, 1024 elems each
  int tid = threadIdx.x;
  const float* p = pred + (size_t)beam * VOCAB + chunk * 1024;
  __shared__ float red[256];
  float v0 = p[tid], v1 = p[tid + 256], v2 = p[tid + 512], v3 = p[tid + 768];
  red[tid] = fmaxf(fmaxf(v0, v1), fmaxf(v2, v3)); __syncthreads();
  for (int s = 128; s > 0; s >>= 1) { if (tid < s) red[tid] = fmaxf(red[tid], red[tid + s]); __syncthreads(); }
  float bm = red[0]; __syncthreads();
  red[tid] = expf(v0 - bm) + expf(v1 - bm) + expf(v2 - bm) + expf(v3 - bm); __syncthreads();
  for (int s = 128; s > 0; s >>= 1) { if (tid < s) red[tid] += red[tid + s]; __syncthreads(); }
  if (tid == 0) { pmax[blockIdx.x] = bm; psum[blockIdx.x] = red[0]; }
}

__global__ void k_lse_fin(const float* __restrict__ pmax, const float* __restrict__ psum,
                          const float* __restrict__ scores, float* __restrict__ scb) {
  int b = threadIdx.x;                                   // 32 threads
  if (b >= BEAM) return;
  float M = -3.0e38f;
  for (int p = 0; p < 125; ++p) M = fmaxf(M, pmax[b * 125 + p]);
  float S = 0.0f;
  for (int p = 0; p < 125; ++p) S += psum[b * 125 + p] * expf(pmax[b * 125 + p] - M);
  scb[b] = scores[b] - (M + logf(S));                    // sc[b,v] = scb[b] + pred[b,v]
}

// ---------------- top-k stage 1: per-block top-32 of sc over 8192 flat entries ----------------
__global__ void k_topk1(const float* __restrict__ pred, const float* __restrict__ scb,
                        int step, float* __restrict__ cval, int* __restrict__ cidx) {
  const int tid = threadIdx.x;
  const int base = blockIdx.x * 8192;                    // 500 blocks * 8192 = 4,096,000
  float vals[32];
#pragma unroll
  for (int k = 0; k < 32; ++k) {
    int flat = base + k * 256 + tid;
    int b = flat / VOCAB;
    float v = pred[flat] + scb[b];
    if (step == 0 && b > 0) v = -1.0e9f;                 // step-0: only beam 0 alive
    vals[k] = v;
  }
  __shared__ float rv[256];
  __shared__ int ri[256];
  for (int r = 0; r < 32; ++r) {
    float bv = -3.0e38f; int bk = 0;
#pragma unroll
    for (int k = 0; k < 32; ++k) if (vals[k] > bv) { bv = vals[k]; bk = k; }
    int bflat = base + bk * 256 + tid;
    rv[tid] = bv; ri[tid] = bflat; __syncthreads();
    for (int s = 128; s > 0; s >>= 1) {
      if (tid < s) {
        if (rv[tid + s] > rv[tid] || (rv[tid + s] == rv[tid] && ri[tid + s] < ri[tid])) {
          rv[tid] = rv[tid + s]; ri[tid] = ri[tid + s];
        }
      }
      __syncthreads();
    }
    if (tid == 0) { cval[blockIdx.x * 32 + r] = rv[0]; cidx[blockIdx.x * 32 + r] = ri[0]; }
    int win = ri[0];
    __syncthreads();
    if (win == bflat) vals[bk] = -3.0e38f;
    __syncthreads();
  }
}

// ---------------- top-k stage 2 + beam state update (single block) ----------------
__global__ void k_update(float* __restrict__ cval, const int* __restrict__ cidx,
                         const float* __restrict__ stateMid, float* __restrict__ state0,
                         float* __restrict__ scores, int* __restrict__ prevw,
                         int* __restrict__ seq, int step) {
  const int tid = threadIdx.x;
  const int NC = 500 * 32;
  volatile float* cv = cval;
  __shared__ float rv[256];
  __shared__ int ri[256];
  __shared__ float topv[BEAM];
  __shared__ int topi[BEAM];
  for (int r = 0; r < BEAM; ++r) {
    float bv = -3.0e38f; int bp = 0;
    for (int t = tid; t < NC; t += 256) {
      float v = cv[t];
      if (v > bv) { bv = v; bp = t; }
    }
    rv[tid] = bv; ri[tid] = bp; __syncthreads();
    for (int s = 128; s > 0; s >>= 1) {
      if (tid < s) {
        if (rv[tid + s] > rv[tid] || (rv[tid + s] == rv[tid] && ri[tid + s] < ri[tid])) {
          rv[tid] = rv[tid + s]; ri[tid] = ri[tid + s];
        }
      }
      __syncthreads();
    }
    if (tid == 0) {
      int p = ri[0];
      topv[r] = rv[0]; topi[r] = cidx[p];
      cval[p] = -3.0e38f;
      __threadfence();
    }
    __syncthreads();
  }
  // beam bookkeeping
  __shared__ int prevs[BEAM], words[BEAM];
  if (tid < BEAM) {
    int flat = topi[tid];
    int prev = flat / VOCAB;
    prevs[tid] = prev;
    words[tid] = flat - prev * VOCAB;
    scores[tid] = topv[tid];
    prevw[tid] = words[tid] = flat - prev * VOCAB;
  }
  __syncthreads();
  // gather h1,c1,h2,c2,ctx (contiguous: 5 * 32 * 128) mid -> carry
  for (int idx = tid; idx < 5 * BEAM * HDIM; idx += 256) {
    int arr = idx >> 12;            // /4096
    int rem = idx & 4095;
    int r = rem >> 7, q = rem & 127;
    state0[arr * 4096 + r * 128 + q] = stateMid[arr * 4096 + prevs[r] * 128 + q];
  }
  // gather + extend sequences
  __shared__ int nseq[BEAM * (NSTEPS + 1)];
  for (int idx = tid; idx < BEAM * (NSTEPS + 1); idx += 256) {
    int r = idx / (NSTEPS + 1), c = idx - r * (NSTEPS + 1);
    nseq[idx] = seq[prevs[r] * (NSTEPS + 1) + c];
  }
  __syncthreads();
  if (tid < BEAM) nseq[tid * (NSTEPS + 1) + step + 1] = words[tid];
  __syncthreads();
  for (int idx = tid; idx < BEAM * (NSTEPS + 1); idx += 256) seq[idx] = nseq[idx];
}

__global__ void k_finalize(const int* __restrict__ seq, const float* __restrict__ scores,
                           float* __restrict__ out) {
  int tid = threadIdx.x;
  for (int i = tid; i < BEAM * (NSTEPS + 1); i += 256) out[i] = (float)seq[i];
  if (tid < BEAM) out[BEAM * (NSTEPS + 1) + tid] = scores[tid];
}

// ---------------- host ----------------
extern "C" void kernel_launch(void* const* d_in, const int* in_sizes, int n_in,
                              void* d_out, int out_size, void* d_ws, size_t ws_size,
                              hipStream_t stream) {
  const float* enc_key = (const float*)d_in[0];
  const float* enc_val = (const float*)d_in[1];
  const float* mask    = (const float*)d_in[2];
  const float* emb     = (const float*)d_in[3];
  const float* Wih1 = (const float*)d_in[4];
  const float* Whh1 = (const float*)d_in[5];
  const float* bih1 = (const float*)d_in[6];
  const float* bhh1 = (const float*)d_in[7];
  const float* Wih2 = (const float*)d_in[8];
  const float* Whh2 = (const float*)d_in[9];
  const float* bih2 = (const float*)d_in[10];
  const float* bhh2 = (const float*)d_in[11];
  const float* Wq  = (const float*)d_in[12];
  const float* bq  = (const float*)d_in[13];
  const float* Wcp = (const float*)d_in[14];
  const float* bcp = (const float*)d_in[15];

  char* ws = (char*)d_ws;
  size_t off = 0;
  auto alloc = [&](size_t bytes) { size_t o = off; off = (off + bytes + 255) & ~(size_t)255; return o; };
  size_t o_wcp   = alloc((size_t)VOCAB * 256 * 2);        // bf16 weights (65.5 MB, L2-resident)
  size_t o_pred  = alloc((size_t)BEAM * VOCAB * 4);       // 16.4 MB
  size_t o_x1    = alloc(BEAM * 256 * 4);
  size_t o_st0   = alloc(5 * BEAM * HDIM * 4);            // carry: h1,c1,h2,c2,ctx
  size_t o_st1   = alloc(5 * BEAM * HDIM * 4);            // mid (this step, pre-gather)
  size_t o_sc    = alloc(BEAM * 4);
  size_t o_pw    = alloc(BEAM * 4);
  size_t o_seq   = alloc(BEAM * (NSTEPS + 1) * 4);
  size_t o_q     = alloc(BEAM * 128 * 4);
  size_t o_e     = alloc(BEAM * TLEN * 4);
  size_t o_xbf   = alloc(BEAM * 256 * 2);
  size_t o_pmax  = alloc(4000 * 4);
  size_t o_psum  = alloc(4000 * 4);
  size_t o_scb   = alloc(BEAM * 4);
  size_t o_cval  = alloc(500 * 32 * 4);
  size_t o_cidx  = alloc(500 * 32 * 4);
  (void)ws_size;

  unsigned short* wcp_bf = (unsigned short*)(ws + o_wcp);
  float* pred   = (float*)(ws + o_pred);
  float* x1     = (float*)(ws + o_x1);
  float* st0    = (float*)(ws + o_st0);
  float* st1    = (float*)(ws + o_st1);
  float* scores = (float*)(ws + o_sc);
  int*   prevw  = (int*)(ws + o_pw);
  int*   seq    = (int*)(ws + o_seq);
  float* query  = (float*)(ws + o_q);
  float* energy = (float*)(ws + o_e);
  unsigned short* xbf = (unsigned short*)(ws + o_xbf);
  float* pmax = (float*)(ws + o_pmax);
  float* psum = (float*)(ws + o_psum);
  float* scb  = (float*)(ws + o_scb);
  float* cval = (float*)(ws + o_cval);
  int*   cidx = (int*)(ws + o_cidx);

  float* h1_0 = st0;               float* c1_0 = st0 + 4096;
  float* h2_0 = st0 + 2 * 4096;    float* c2_0 = st0 + 3 * 4096;
  float* ctx0 = st0 + 4 * 4096;
  float* h1_1 = st1;               float* c1_1 = st1 + 4096;
  float* h2_1 = st1 + 2 * 4096;    float* c2_1 = st1 + 3 * 4096;
  float* ctx1 = st1 + 4 * 4096;

  k_convert_wcp<<<32000, 256, 0, stream>>>(Wcp, wcp_bf);
  k_init<<<1, 256, 0, stream>>>(st0, scores, prevw, seq);

  for (int i = 0; i < NSTEPS; ++i) {
    k_build_x1<<<32, 256, 0, stream>>>(prevw, emb, ctx0, x1);
    k_lstm<<<16, 256, 0, stream>>>(x1, 256, h1_0, c1_0, Wih1, Whh1, bih1, bhh1, h1_1, c1_1);
    k_lstm<<<16, 256, 0, stream>>>(h1_1, 128, h2_0, c2_0, Wih2, Whh2, bih2, bhh2, h2_1, c2_1);
    k_query<<<16, 256, 0, stream>>>(h2_1, Wq, bq, query);
    k_energy<<<64, 256, 0, stream>>>(query, enc_key, energy);
    k_attn<<<32, 256, 0, stream>>>(energy, mask, enc_val, h2_1, ctx1, xbf);
    k_pred_wmma<<<2000, 256, 0, stream>>>(xbf, wcp_bf, bcp, pred);
    k_lse_part<<<4000, 256, 0, stream>>>(pred, pmax, psum);
    k_lse_fin<<<1, 32, 0, stream>>>(pmax, psum, scores, scb);
    k_topk1<<<500, 256, 0, stream>>>(pred, scb, i, cval, cidx);
    k_update<<<1, 256, 0, stream>>>(cval, cidx, st1, st0, scores, prevw, seq, i);
  }
  k_finalize<<<1, 256, 0, stream>>>(seq, scores, (float*)d_out);
}
